// Model_824633721730
// MI455X (gfx1250) — matrix operand order repaired and checked by reading, not compiled, not run on hardware
//
#include <hip/hip_runtime.h>
#include <hip/hip_bf16.h>

#define HDIM 64
#define NS_N 40000
#define NV_N 20000
#define E_N  1000000
#define L_N  200000
#define FS_K 2613
#define FV_K 4517
#define BN_EPS 1e-5f

typedef __attribute__((ext_vector_type(16))) _Float16 v16h;
typedef __attribute__((ext_vector_type(8)))  float    v8f;

#define WMMA_F16(a, b, c) \
    __builtin_amdgcn_wmma_f32_16x16x32_f16(false, (a), false, (b), (short)0, (c), false, false)

// A-fragment element j -> k offset within the 32-wide k-step (ISA 7.12.2 16-bit A 16x32)
__device__ __forceinline__ int a_koff(int j, int kbase) {
    int i = j >> 1;                                   // VGPR index 0..7
    return kbase + 2 * i + ((i >= 4) ? 8 : 0) + (j & 1);
}

// ---------------------------------------------------------------------------
// out[N x 64] = X[N x K] @ W[64 x K]^T + bias + emb          (WMMA f16->f32)
// wave -> 16 rows x 64 cols (A frag reused across 4 B tiles); block = 8 waves
// All raw loads of a k-step are batched before conversion -> one wait, deep MLP.
// ---------------------------------------------------------------------------
__global__ void proj_wmma(const float* __restrict__ X, const float* __restrict__ W,
                          const float* __restrict__ bias, const float* __restrict__ emb,
                          float* __restrict__ out, int N, int K) {
    const int lane = threadIdx.x & 31, wave = threadIdx.x >> 5;
    const int row0 = (blockIdx.x * 8 + wave) * 16;
    int mA = row0 + (lane & 15);
    if (mA > N - 1) mA = N - 1;                 // clamp for safe loads; stores guarded
    const int kbA = (lane < 16) ? 0 : 8;
    const int kbB = (lane < 16) ? 0 : 16;

    const float* __restrict__ arow = X + (size_t)mA * K;
    const float* wr[4];
#pragma unroll
    for (int t = 0; t < 4; ++t)
        wr[t] = W + (size_t)(t * 16 + (lane & 15)) * K;

    v8f c[4] = {{}, {}, {}, {}};
    const int Kmain = K & ~31;
    for (int k0 = 0; k0 < Kmain; k0 += 32) {
        // ---- batched raw loads (no consumer between them) ----
        float ar[16];
#pragma unroll
        for (int j = 0; j < 16; ++j) ar[j] = arow[k0 + kbB + j];
        float br[4][16];
#pragma unroll
        for (int t = 0; t < 4; ++t)
#pragma unroll
            for (int i = 0; i < 16; ++i) br[t][i] = wr[t][k0 + kbB + i];
        __builtin_prefetch(arow + k0 + 64, 0, 1);   // global_prefetch next A chunk
        // ---- convert ----
        v16h a;
#pragma unroll
        for (int j = 0; j < 16; ++j) a[j] = (_Float16)ar[j];
        v16h b[4];
#pragma unroll
        for (int t = 0; t < 4; ++t)
#pragma unroll
            for (int i = 0; i < 16; ++i) b[t][i] = (_Float16)br[t][i];
        // ---- matrix ops ----
#pragma unroll
        for (int t = 0; t < 4; ++t) c[t] = WMMA_F16(a, b[t], c[t]);
    }
    if (Kmain < K) {                            // single branch-free tail step
        v16h a;
#pragma unroll
        for (int j = 0; j < 16; ++j) {
            int ko = Kmain + kbB + j;
            float v = arow[min(ko, K - 1)];
            a[j] = (_Float16)((ko < K) ? v : 0.0f);
        }
        v16h b[4];
#pragma unroll
        for (int t = 0; t < 4; ++t)
#pragma unroll
            for (int i = 0; i < 16; ++i) {
                int ko = Kmain + kbB + i;
                float v = wr[t][min(ko, K - 1)];
                b[t][i] = (_Float16)((ko < K) ? v : 0.0f);
            }
#pragma unroll
        for (int t = 0; t < 4; ++t) c[t] = WMMA_F16(a, b[t], c[t]);
    }

    const int moff = (lane < 16) ? 0 : 8;
#pragma unroll
    for (int t = 0; t < 4; ++t) {
        int n = t * 16 + (lane & 15);
        float bn = bias[n];
#pragma unroll
        for (int r = 0; r < 8; ++r) {
            int row = row0 + moff + r;
            if (row < N)
                out[(size_t)row * HDIM + n] = c[t][r] + bn + emb[(size_t)row * HDIM + n];
        }
    }
}

// ---------------------------------------------------------------------------
// Fused SAGE combine: out = (agg/max(cnt,1)) @ Wl^T + bl + xdst @ Wr^T  (K=64)
// wave -> 16 rows x 64 cols; rows 256B-aligned -> float4 batched loads
// ---------------------------------------------------------------------------
__global__ void sage_combine_wmma(const float* __restrict__ agg, const float* __restrict__ cnt,
                                  const float* __restrict__ xdst,
                                  const float* __restrict__ Wl, const float* __restrict__ bl,
                                  const float* __restrict__ Wr,
                                  float* __restrict__ out, int N) {
    const int lane = threadIdx.x & 31, wave = threadIdx.x >> 5;
    const int row0 = (blockIdx.x * 8 + wave) * 16;
    int mA = row0 + (lane & 15);
    if (mA > N - 1) mA = N - 1;
    const int kbA = (lane < 16) ? 0 : 8;
    const int kbB = (lane < 16) ? 0 : 16;

    const float scale = 1.0f / fmaxf(cnt[mA], 1.0f);
    const float* __restrict__ arow = agg  + (size_t)mA * HDIM;
    const float* __restrict__ xrow = xdst + (size_t)mA * HDIM;

    v8f c[4] = {{}, {}, {}, {}};
#pragma unroll
    for (int pass = 0; pass < 2; ++pass) {
        const float* __restrict__ asrc = pass ? xrow : arow;
        const float* __restrict__ wsrc = pass ? Wr : Wl;
        const float  asc = pass ? 1.0f : scale;
#pragma unroll
        for (int k0 = 0; k0 < HDIM; k0 += 32) {
            // ---- batched raw float4 loads ----
            float4 ar[4];
            ar[0] = *(const float4*)(asrc + k0 + kbB);
            ar[1] = *(const float4*)(asrc + k0 + kbB + 4);
            ar[2] = *(const float4*)(asrc + k0 + kbB + 8);
            ar[3] = *(const float4*)(asrc + k0 + kbB + 12);
            float4 br[4][4];
#pragma unroll
            for (int t = 0; t < 4; ++t) {
                const float* w = wsrc + (size_t)(t * 16 + (lane & 15)) * HDIM + k0 + kbB;
#pragma unroll
                for (int i = 0; i < 4; ++i) br[t][i] = ((const float4*)w)[i];
            }
            // ---- convert ----
            v16h a;
#pragma unroll
            for (int i = 0; i < 4; ++i) {
                a[4 * i + 0] = (_Float16)(ar[i].x * asc);
                a[4 * i + 1] = (_Float16)(ar[i].y * asc);
                a[4 * i + 2] = (_Float16)(ar[i].z * asc);
                a[4 * i + 3] = (_Float16)(ar[i].w * asc);
            }
            v16h b[4];
#pragma unroll
            for (int t = 0; t < 4; ++t)
#pragma unroll
                for (int i = 0; i < 4; ++i) {
                    b[t][4 * i + 0] = (_Float16)br[t][i].x;
                    b[t][4 * i + 1] = (_Float16)br[t][i].y;
                    b[t][4 * i + 2] = (_Float16)br[t][i].z;
                    b[t][4 * i + 3] = (_Float16)br[t][i].w;
                }
            // ---- matrix ops ----
#pragma unroll
            for (int t = 0; t < 4; ++t) c[t] = WMMA_F16(a, b[t], c[t]);
        }
    }

    const int moff = (lane < 16) ? 0 : 8;
#pragma unroll
    for (int t = 0; t < 4; ++t) {
        int n = t * 16 + (lane & 15);
        float bn = bl[n];
#pragma unroll
        for (int r = 0; r < 8; ++r) {
            int row = row0 + moff + r;
            if (row < N) out[(size_t)row * HDIM + n] = c[t][r] + bn;
        }
    }
}

// ---------------------------------------------------------------------------
__global__ void edge_count(const int* __restrict__ dst, int E, float* __restrict__ cnt) {
    int e = blockIdx.x * blockDim.x + threadIdx.x;
    if (e < E) unsafeAtomicAdd(&cnt[dst[e]], 1.0f);
}

// thread = (edge, 4 features): float4 gather + 4 hardware f32 atomics
__global__ void edge_scatter(const float* __restrict__ hsrc,
                             const int* __restrict__ src, const int* __restrict__ dst,
                             int E, float* __restrict__ agg) {
    int tid = blockIdx.x * blockDim.x + threadIdx.x;
    int e = tid >> 4, f = (tid & 15) * 4;
    if (e < E) {
        const float4 v = *(const float4*)(hsrc + (size_t)src[e] * HDIM + f);
        float* p = agg + (size_t)dst[e] * HDIM + f;
        unsafeAtomicAdd(p + 0, v.x);
        unsafeAtomicAdd(p + 1, v.y);
        unsafeAtomicAdd(p + 2, v.z);
        unsafeAtomicAdd(p + 3, v.w);
    }
}

__global__ void bn_stats(const float* __restrict__ x, int N, float* __restrict__ stats) {
    __shared__ float lds[256];
    const int t = threadIdx.x, f = t & 63, q = t >> 6;
    float s = 0.0f, ss = 0.0f;
    for (int row = blockIdx.x * 4 + q; row < N; row += gridDim.x * 4) {
        float v = x[(size_t)row * HDIM + f];
        s += v; ss += v * v;
    }
    lds[t] = s; __syncthreads();
    if (q == 0) unsafeAtomicAdd(&stats[f], lds[f] + lds[f + 64] + lds[f + 128] + lds[f + 192]);
    __syncthreads();
    lds[t] = ss; __syncthreads();
    if (q == 0) unsafeAtomicAdd(&stats[64 + f], lds[f] + lds[f + 64] + lds[f + 128] + lds[f + 192]);
}

__global__ void bn_apply_relu(float* __restrict__ x, const float* __restrict__ stats,
                              const float* __restrict__ g, const float* __restrict__ b,
                              int N) {
    int tid = blockIdx.x * blockDim.x + threadIdx.x;
    if (tid < N * HDIM) {
        int f = tid & 63;
        float invN = 1.0f / (float)N;
        float m = stats[f] * invN;
        float var = stats[64 + f] * invN - m * m;
        float y = (x[tid] - m) * rsqrtf(var + BN_EPS) * g[f] + b[f];
        x[tid] = fmaxf(y, 0.0f);
    }
}

__global__ void classify(const float* __restrict__ st3, const float* __restrict__ inv3,
                         const int* __restrict__ esrc, const int* __restrict__ edst,
                         float* __restrict__ pred, int L) {
    const int lane = threadIdx.x & 31, wave = threadIdx.x >> 5;
    const int e = blockIdx.x * 8 + wave;
    if (e >= L) return;
    const float* a = st3  + (size_t)esrc[e] * HDIM;
    const float* b = inv3 + (size_t)edst[e] * HDIM;
    float s = a[lane] * b[lane] + a[lane + 32] * b[lane + 32];
#pragma unroll
    for (int m = 16; m >= 1; m >>= 1) s += __shfl_xor(s, m, 32);
    if (lane == 0) pred[e] = s;
}

// ---------------------------------------------------------------------------
extern "C" void kernel_launch(void* const* d_in, const int* in_sizes, int n_in,
                              void* d_out, int out_size, void* d_ws, size_t ws_size,
                              hipStream_t stream) {
    (void)in_sizes; (void)n_in; (void)out_size; (void)ws_size;

    const float* x_st  = (const float*)d_in[0];
    const float* x_vc  = (const float*)d_in[1];
    const int* src_s2i = (const int*)d_in[4];
    const int* dst_s2i = (const int*)d_in[5];
    const int* src_i2s = (const int*)d_in[6];
    const int* dst_i2s = (const int*)d_in[7];
    const int* ell_src = (const int*)d_in[8];
    const int* ell_dst = (const int*)d_in[9];
    const float* st_W  = (const float*)d_in[10];
    const float* st_b  = (const float*)d_in[11];
    const float* vc_W  = (const float*)d_in[12];
    const float* vc_b  = (const float*)d_in[13];
    const float* st_emb = (const float*)d_in[14];
    const float* vc_emb = (const float*)d_in[15];
    const float* c_Wl[6]; const float* c_bl[6]; const float* c_Wr[6];
    for (int i = 0; i < 6; ++i) {           // order: 1_s2i, 1_i2s, 2_s2i, 2_i2s, 3_s2i, 3_i2s
        c_Wl[i] = (const float*)d_in[16 + 3 * i + 0];
        c_bl[i] = (const float*)d_in[16 + 3 * i + 1];
        c_Wr[i] = (const float*)d_in[16 + 3 * i + 2];
    }
    const float* bn1_st_g  = (const float*)d_in[34]; const float* bn1_st_b  = (const float*)d_in[35];
    const float* bn1_inv_g = (const float*)d_in[36]; const float* bn1_inv_b = (const float*)d_in[37];
    const float* bn2_st_g  = (const float*)d_in[38]; const float* bn2_st_b  = (const float*)d_in[39];
    const float* bn2_inv_g = (const float*)d_in[40]; const float* bn2_inv_b = (const float*)d_in[41];

    // ---- workspace carve (all matrix regions 256B-aligned) ----
    float* ws = (float*)d_ws;
    float* h_st    = ws;                    ws += (size_t)NS_N * HDIM;
    float* h_inv   = ws;                    ws += (size_t)NV_N * HDIM;
    float* stA     = ws;                    ws += (size_t)NS_N * HDIM;
    float* invA    = ws;                    ws += (size_t)NV_N * HDIM;
    float* stB     = ws;                    ws += (size_t)NS_N * HDIM;
    float* invB    = ws;                    ws += (size_t)NV_N * HDIM;
    float* agg_st  = ws;                    ws += (size_t)NS_N * HDIM;
    float* agg_inv = ws;                    ws += (size_t)NV_N * HDIM;
    float* cnt_st  = ws;                    ws += NS_N;
    float* cnt_inv = ws;                    ws += NV_N;
    float* stats   = ws;                    ws += 128;

    float* pred     = (float*)d_out;
    float* out_st3  = pred + L_N;
    float* out_inv3 = out_st3 + (size_t)NS_N * HDIM;

    const int TB = 256;
    const dim3 blk(TB);
    const int gProjSt = (NS_N + 127) / 128, gProjVc = (NV_N + 127) / 128;
    const int gEdge   = (E_N + TB - 1) / TB;
    const int gScat   = (int)(((size_t)E_N * 16 + TB - 1) / TB);
    const int gElSt   = (NS_N * HDIM + TB - 1) / TB, gElVc = (NV_N * HDIM + TB - 1) / TB;
    const int gCls    = (L_N + 7) / 8;

    // ---- degree counts (identical edge lists across layers -> once) ----
    hipMemsetAsync(cnt_st,  0, NS_N * sizeof(float), stream);
    hipMemsetAsync(cnt_inv, 0, NV_N * sizeof(float), stream);
    edge_count<<<gEdge, blk, 0, stream>>>(dst_s2i, E_N, cnt_inv);
    edge_count<<<gEdge, blk, 0, stream>>>(dst_i2s, E_N, cnt_st);

    // ---- input projections (WMMA) ----
    proj_wmma<<<gProjSt, blk, 0, stream>>>(x_st, st_W, st_b, st_emb, h_st, NS_N, FS_K);
    proj_wmma<<<gProjVc, blk, 0, stream>>>(x_vc, vc_W, vc_b, vc_emb, h_inv, NV_N, FV_K);

    // ---- layer 1 ----
    hipMemsetAsync(agg_inv, 0, (size_t)NV_N * HDIM * sizeof(float), stream);
    edge_scatter<<<gScat, blk, 0, stream>>>(h_st, src_s2i, dst_s2i, E_N, agg_inv);
    sage_combine_wmma<<<gProjVc, blk, 0, stream>>>(agg_inv, cnt_inv, h_inv,
                                                   c_Wl[0], c_bl[0], c_Wr[0], invA, NV_N);
    hipMemsetAsync(agg_st, 0, (size_t)NS_N * HDIM * sizeof(float), stream);
    edge_scatter<<<gScat, blk, 0, stream>>>(h_inv, src_i2s, dst_i2s, E_N, agg_st);
    sage_combine_wmma<<<gProjSt, blk, 0, stream>>>(agg_st, cnt_st, h_st,
                                                   c_Wl[1], c_bl[1], c_Wr[1], stA, NS_N);
    hipMemsetAsync(stats, 0, 128 * sizeof(float), stream);
    bn_stats<<<256, blk, 0, stream>>>(invA, NV_N, stats);
    bn_apply_relu<<<gElVc, blk, 0, stream>>>(invA, stats, bn1_inv_g, bn1_inv_b, NV_N);
    hipMemsetAsync(stats, 0, 128 * sizeof(float), stream);
    bn_stats<<<256, blk, 0, stream>>>(stA, NS_N, stats);
    bn_apply_relu<<<gElSt, blk, 0, stream>>>(stA, stats, bn1_st_g, bn1_st_b, NS_N);

    // ---- layer 2 ----
    hipMemsetAsync(agg_inv, 0, (size_t)NV_N * HDIM * sizeof(float), stream);
    edge_scatter<<<gScat, blk, 0, stream>>>(stA, src_s2i, dst_s2i, E_N, agg_inv);
    sage_combine_wmma<<<gProjVc, blk, 0, stream>>>(agg_inv, cnt_inv, invA,
                                                   c_Wl[2], c_bl[2], c_Wr[2], invB, NV_N);
    hipMemsetAsync(agg_st, 0, (size_t)NS_N * HDIM * sizeof(float), stream);
    edge_scatter<<<gScat, blk, 0, stream>>>(invA, src_i2s, dst_i2s, E_N, agg_st);
    sage_combine_wmma<<<gProjSt, blk, 0, stream>>>(agg_st, cnt_st, stA,
                                                   c_Wl[3], c_bl[3], c_Wr[3], stB, NS_N);
    hipMemsetAsync(stats, 0, 128 * sizeof(float), stream);
    bn_stats<<<256, blk, 0, stream>>>(invB, NV_N, stats);
    bn_apply_relu<<<gElVc, blk, 0, stream>>>(invB, stats, bn2_inv_g, bn2_inv_b, NV_N);
    hipMemsetAsync(stats, 0, 128 * sizeof(float), stream);
    bn_stats<<<256, blk, 0, stream>>>(stB, NS_N, stats);
    bn_apply_relu<<<gElSt, blk, 0, stream>>>(stB, stats, bn2_st_g, bn2_st_b, NS_N);

    // ---- layer 3 (no BN), write directly to d_out ----
    hipMemsetAsync(agg_inv, 0, (size_t)NV_N * HDIM * sizeof(float), stream);
    edge_scatter<<<gScat, blk, 0, stream>>>(stB, src_s2i, dst_s2i, E_N, agg_inv);
    sage_combine_wmma<<<gProjVc, blk, 0, stream>>>(agg_inv, cnt_inv, invB,
                                                   c_Wl[4], c_bl[4], c_Wr[4], out_inv3, NV_N);
    hipMemsetAsync(agg_st, 0, (size_t)NS_N * HDIM * sizeof(float), stream);
    edge_scatter<<<gScat, blk, 0, stream>>>(invB, src_i2s, dst_i2s, E_N, agg_st);
    sage_combine_wmma<<<gProjSt, blk, 0, stream>>>(agg_st, cnt_st, stB,
                                                   c_Wl[5], c_bl[5], c_Wr[5], out_st3, NS_N);

    // ---- classifier ----
    classify<<<gCls, blk, 0, stream>>>(out_st3, out_inv3, ell_src, ell_dst, pred, L_N);
}
